// GCNmass_79121887527626
// MI455X (gfx1250) — matrix operand order repaired and checked by reading, not compiled, not run on hardware
//
#include <hip/hip_runtime.h>
#include <hip/hip_bf16.h>
#include <math.h>

typedef _Float16 half_t;
typedef __attribute__((ext_vector_type(16))) _Float16 v16h;
typedef __attribute__((ext_vector_type(8)))  float    v8f;

// ---------------------------------------------------------------------------
// Degree count: degf[dst[e]] += 1.0
// ---------------------------------------------------------------------------
__global__ void deg_count_kernel(const int* __restrict__ dst,
                                 float* __restrict__ degf, int E) {
    int t = blockIdx.x * blockDim.x + threadIdx.x;
    if (t < E) atomicAdd(&degf[dst[t]], 1.0f);
}

// dinv[n] = rsqrt(deg[n] + 1)  (self-loop included)
__global__ void dinv_kernel(const float* __restrict__ degf,
                            float* __restrict__ dinv, int N) {
    int n = blockIdx.x * blockDim.x + threadIdx.x;
    if (n < N) dinv[n] = rsqrtf(degf[n] + 1.0f);
}

// ---------------------------------------------------------------------------
// GEMM1: g1[n, 0:64] = (x[n, 0:64] @ W1[64, 64]) * dinv[n]
// One wave = one 16x16 tile, K = 64 via two v_wmma_f32_16x16x32_f16.
// ---------------------------------------------------------------------------
__global__ __launch_bounds__(256) void gemm1_wmma_kernel(
    const float* __restrict__ x, const float* __restrict__ W,
    const float* __restrict__ dinv, float* __restrict__ g, int nRowTiles) {
    const int lane    = threadIdx.x & 31;
    const int wave    = blockIdx.x * 8 + (threadIdx.x >> 5);
    const int rowTile = wave >> 2;     // 4 column tiles (64 cols)
    const int colTile = wave & 3;
    if (rowTile >= nRowTiles) return;  // wave-uniform: EXEC stays all-1s

    const int halfSel = (lane >> 4) & 1;     // 0: lanes 0-15, 1: lanes 16-31
    const int r       = rowTile * 16 + (lane & 15);
    const float* xr   = x + (size_t)r * 64 + (halfSel ? 8 : 0);

    // A tiles (16x32 f16 per WMMA): elems 0..7 -> K base+{0..7}, 8..15 -> K base+16+{0..7}
    v16h a0, a1;
#pragma unroll
    for (int i = 0; i < 8; ++i) {
        a0[i]     = (half_t)xr[i];
        a0[8 + i] = (half_t)xr[16 + i];
        a1[i]     = (half_t)xr[32 + i];
        a1[8 + i] = (half_t)xr[48 + i];
    }

    // B tiles (32x16 f16): lane holds column n, elems 0..7 -> K kb+{0..7}, 8..15 -> K kb+16+{0..7}
    const int n  = colTile * 16 + (lane & 15);
    const int kb = halfSel ? 8 : 0;
    v16h b0, b1;
#pragma unroll
    for (int i = 0; i < 16; ++i) {
        b0[i] = (half_t)W[(kb + i + (i & 8)) * 64 + n];
        b1[i] = (half_t)W[(kb + 32 + i + (i & 8)) * 64 + n];
    }

    v8f c = {};
    c = __builtin_amdgcn_wmma_f32_16x16x32_f16(false, a0, false, b0, (short)0, c, false, false);
    c = __builtin_amdgcn_wmma_f32_16x16x32_f16(false, a1, false, b1, (short)0, c, false, false);

    // D layout: VGPR rr -> row rr + 8*halfSel ; epilogue: scale by dinv[row]
#pragma unroll
    for (int rr = 0; rr < 8; ++rr) {
        const int m = rowTile * 16 + rr + (halfSel ? 8 : 0);
        g[(size_t)m * 64 + n] = c[rr] * dinv[m];
    }
}

// ---------------------------------------------------------------------------
// GEMM2: g2[n, 0:36] = (h1[n, 0:64](f16) @ W2[64, 33]) * dinv[n]
// 3 column tiles (48 cols); cols >= 33 masked to zero in B; store cols < 36
// (stride 36 so the generic float4 scatter sees exact zeros in pad columns).
// ---------------------------------------------------------------------------
__global__ __launch_bounds__(256) void gemm2_wmma_kernel(
    const half_t* __restrict__ h, const float* __restrict__ W2,
    const float* __restrict__ dinv, float* __restrict__ g2, int nRowTiles) {
    const int lane    = threadIdx.x & 31;
    const int wave    = blockIdx.x * 8 + (threadIdx.x >> 5);
    const int rowTile = wave / 3;
    const int colTile = wave % 3;
    if (rowTile >= nRowTiles) return;  // wave-uniform

    const int halfSel = (lane >> 4) & 1;
    const int r       = rowTile * 16 + (lane & 15);
    const half_t* hr  = h + (size_t)r * 64 + (halfSel ? 8 : 0);

    v16h a0, a1;
#pragma unroll
    for (int i = 0; i < 8; ++i) {
        a0[i]     = hr[i];
        a0[8 + i] = hr[16 + i];
        a1[i]     = hr[32 + i];
        a1[8 + i] = hr[48 + i];
    }

    const int n  = colTile * 16 + (lane & 15);
    const int kb = halfSel ? 8 : 0;
    v16h b0, b1;
#pragma unroll
    for (int i = 0; i < 16; ++i) {
        b0[i] = (n < 33) ? (half_t)W2[(kb + i + (i & 8)) * 33 + n]      : (half_t)0.0f;
        b1[i] = (n < 33) ? (half_t)W2[(kb + 32 + i + (i & 8)) * 33 + n] : (half_t)0.0f;
    }

    v8f c = {};
    c = __builtin_amdgcn_wmma_f32_16x16x32_f16(false, a0, false, b0, (short)0, c, false, false);
    c = __builtin_amdgcn_wmma_f32_16x16x32_f16(false, a1, false, b1, (short)0, c, false, false);

#pragma unroll
    for (int rr = 0; rr < 8; ++rr) {
        const int m = rowTile * 16 + rr + (halfSel ? 8 : 0);
        if (n < 36) g2[(size_t)m * 36 + n] = c[rr] * dinv[m];
    }
}

// ---------------------------------------------------------------------------
// Edge scatter: acc[dst] += g[src]  (rows pre-scaled by dinv[src]).
// One thread = one float4 chunk of one edge row; reads coalesce within an
// edge, atomics go to L2 (h fits in 192 MB L2).
// ---------------------------------------------------------------------------
__global__ void scatter_add_kernel(const float* __restrict__ g,
                                   float* __restrict__ acc,
                                   const int* __restrict__ src,
                                   const int* __restrict__ dst,
                                   int E, int chunks, int stride) {
    long long t = (long long)blockIdx.x * blockDim.x + threadIdx.x;
    long long total = (long long)E * chunks;
    if (t >= total) return;
    int e  = (int)(t / chunks);
    int ch = (int)(t % chunks);
    int s = src[e], d = dst[e];
    const float4 v = *(const float4*)(g + (size_t)s * stride + ch * 4);
    float* ap = acc + (size_t)d * stride + ch * 4;
    atomicAdd(ap + 0, v.x);
    atomicAdd(ap + 1, v.y);
    atomicAdd(ap + 2, v.z);
    atomicAdd(ap + 3, v.w);
}

// Layer-1 epilogue: h1 = (f16) relu(dinv[n]*(acc + g) + b1)
__global__ void finish1_kernel(const float* __restrict__ g1,
                               const float* __restrict__ acc1,
                               const float* __restrict__ dinv,
                               const float* __restrict__ b1,
                               half_t* __restrict__ h1, int N) {
    long long t = (long long)blockIdx.x * blockDim.x + threadIdx.x;
    if (t >= (long long)N * 64) return;
    int n = (int)(t >> 6);
    int c = (int)(t & 63);
    float v = dinv[n] * (acc1[t] + g1[t]) + b1[c];
    h1[t] = (half_t)fmaxf(v, 0.0f);
}

// Layer-2 epilogue: pos[n,0:32] and mass[n] = exp(col 32), into d_out.
__global__ void finish2_kernel(const float* __restrict__ g2,
                               const float* __restrict__ acc2,
                               const float* __restrict__ dinv,
                               const float* __restrict__ b2,
                               float* __restrict__ out, int N) {
    long long t = (long long)blockIdx.x * blockDim.x + threadIdx.x;
    if (t >= (long long)N * 33) return;
    int n = (int)(t / 33);
    int c = (int)(t % 33);
    size_t idx = (size_t)n * 36 + c;
    float v = dinv[n] * (acc2[idx] + g2[idx]) + b2[c];
    if (c < 32) out[(size_t)n * 32 + c] = v;
    else        out[(size_t)N * 32 + n] = expf(v);
}

// ---------------------------------------------------------------------------
extern "C" void kernel_launch(void* const* d_in, const int* in_sizes, int n_in,
                              void* d_out, int out_size, void* d_ws, size_t ws_size,
                              hipStream_t stream) {
    const float* x  = (const float*)d_in[0];
    const float* W1 = (const float*)d_in[1];
    const float* b1 = (const float*)d_in[2];
    const float* W2 = (const float*)d_in[3];
    const float* b2 = (const float*)d_in[4];
    const int*   ei = (const int*)d_in[5];

    const int N = in_sizes[0] / 64;
    const int E = in_sizes[5] / 2;
    const int* src = ei;
    const int* dst = ei + E;

    // Workspace carve-out (256B aligned); g2/acc2 reuse g1/acc1 storage.
    char* ws = (char*)d_ws;
    size_t off = 0;
    auto carve = [&](size_t bytes) -> void* {
        void* p = ws + off;
        off += (bytes + 255) & ~(size_t)255;
        return p;
    };
    float*  degf = (float*)carve((size_t)N * 4);
    float*  dinv = (float*)carve((size_t)N * 4);
    float*  g1   = (float*)carve((size_t)N * 64 * 4);  // also g2 (N*36*4 fits)
    float*  acc  = (float*)carve((size_t)N * 64 * 4);  // also acc2
    half_t* h1   = (half_t*)carve((size_t)N * 64 * 2);
    float*  g2   = g1;
    float*  acc2 = acc;

    const int nRowTiles = (N + 15) / 16;  // 6250 (N divisible by 16)

    // --- degrees / normalization ---
    hipMemsetAsync(degf, 0, (size_t)N * 4, stream);
    deg_count_kernel<<<(E + 255) / 256, 256, 0, stream>>>(dst, degf, E);
    dinv_kernel<<<(N + 255) / 256, 256, 0, stream>>>(degf, dinv, N);

    // --- layer 1 ---
    gemm1_wmma_kernel<<<(nRowTiles * 4 + 7) / 8, 256, 0, stream>>>(x, W1, dinv, g1, nRowTiles);
    hipMemsetAsync(acc, 0, (size_t)N * 64 * 4, stream);
    {
        long long total = (long long)E * 16;
        scatter_add_kernel<<<(unsigned)((total + 255) / 256), 256, 0, stream>>>(
            g1, acc, src, dst, E, 16, 64);
    }
    {
        long long total = (long long)N * 64;
        finish1_kernel<<<(unsigned)((total + 255) / 256), 256, 0, stream>>>(
            g1, acc, dinv, b1, h1, N);
    }

    // --- layer 2 ---
    gemm2_wmma_kernel<<<(nRowTiles * 3 + 7) / 8, 256, 0, stream>>>(h1, W2, dinv, g2, nRowTiles);
    hipMemsetAsync(acc2, 0, (size_t)N * 36 * 4, stream);
    {
        long long total = (long long)E * 9;
        scatter_add_kernel<<<(unsigned)((total + 255) / 256), 256, 0, stream>>>(
            g2, acc2, src, dst, E, 9, 36);
    }
    {
        long long total = (long long)N * 33;
        finish2_kernel<<<(unsigned)((total + 255) / 256), 256, 0, stream>>>(
            g2, acc2, dinv, b2, (float*)d_out, N);
    }
}
